// CausalAttn_23167053594697
// MI455X (gfx1250) — matrix operand order repaired and checked by reading, not compiled, not run on hardware
//
#include <hip/hip_runtime.h>

#define B_   2
#define T_   2048
#define D_   1024
#define H_   16
#define DH_  64
#define WIN_ 1024
#define M_   (B_ * T_)   // 4096 rows

typedef __bf16 bf16_t;
typedef __attribute__((ext_vector_type(16))) __bf16 v16bf;
typedef __attribute__((ext_vector_type(8)))  __bf16 v8bf;
typedef __attribute__((ext_vector_type(8)))  float  v8f;

__device__ inline v8f wmma_bf16(v16bf a, v16bf b, v8f c) {
  // v_wmma_f32_16x16x32_bf16: D = A(16x32) * B(32x16) + C(16x16 f32)
  return __builtin_amdgcn_wmma_f32_16x16x32_bf16(false, a, false, b,
                                                 (short)0, c, false, false);
}

__device__ inline v16bf cat8(v8bf lo, v8bf hi) {
  v16bf r;
#pragma unroll
  for (int i = 0; i < 8; ++i) { r[i] = lo[i]; r[i + 8] = hi[i]; }
  return r;
}

__device__ inline v8f vzero8() {
  v8f z;
#pragma unroll
  for (int i = 0; i < 8; ++i) z[i] = 0.0f;
  return z;
}

// ---------------- fp32 -> bf16 convert ----------------
__global__ void f2bf_kernel(const float* __restrict__ src,
                            bf16_t* __restrict__ dst, int n) {
  int i = blockIdx.x * blockDim.x + threadIdx.x;
  if (i < n) dst[i] = (bf16_t)src[i];
}

// ---------------- bf16 GEMM: C[m,n] = sum_k A[m,k]*W[n,k] + bias[n] --------
// M=4096, N=1024, K=1024 fixed. One wave computes a 16x64 tile.
// Register double-buffered: next k-step fragments load while current WMMAs run.
__global__ __launch_bounds__(256) void gemm_bf16_kernel(
    const bf16_t* __restrict__ A, const bf16_t* __restrict__ W,
    const float* __restrict__ bias, float* __restrict__ C) {
  const int lane = threadIdx.x & 31;
  const int wv   = threadIdx.x >> 5;
  const int flat = blockIdx.x * 8 + wv;   // 4096 waves total
  const int mt   = flat >> 4;             // 256 m-tiles
  const int n64  = flat & 15;             // 16 n64-tiles
  const int half = lane >> 4, l16 = lane & 15;
  const int mbase = mt * 16, nbase = n64 * 64;

  v8f acc0 = vzero8(), acc1 = vzero8(), acc2 = vzero8(), acc3 = vzero8();
  const bf16_t* arow = A + (size_t)(mbase + l16) * D_ + half * 16;
  const bf16_t* w0 = W + (size_t)(nbase +  0 + l16) * D_ + half * 16;
  const bf16_t* w1 = W + (size_t)(nbase + 16 + l16) * D_ + half * 16;
  const bf16_t* w2 = W + (size_t)(nbase + 32 + l16) * D_ + half * 16;
  const bf16_t* w3 = W + (size_t)(nbase + 48 + l16) * D_ + half * 16;

  // prologue: fragments for k0 = 0
  v16bf af  = cat8(*(const v8bf*)(arow), *(const v8bf*)(arow + 8));
  v16bf bf0 = cat8(*(const v8bf*)(w0),   *(const v8bf*)(w0 + 8));
  v16bf bf1 = cat8(*(const v8bf*)(w1),   *(const v8bf*)(w1 + 8));
  v16bf bf2 = cat8(*(const v8bf*)(w2),   *(const v8bf*)(w2 + 8));
  v16bf bf3 = cat8(*(const v8bf*)(w3),   *(const v8bf*)(w3 + 8));

  for (int k0 = 0; k0 < D_ - 32; k0 += 32) {
    const int kn = k0 + 32;
    // issue next-step loads BEFORE consuming current fragments
    v16bf afn  = cat8(*(const v8bf*)(arow + kn), *(const v8bf*)(arow + kn + 8));
    v16bf b0n  = cat8(*(const v8bf*)(w0 + kn),   *(const v8bf*)(w0 + kn + 8));
    v16bf b1n  = cat8(*(const v8bf*)(w1 + kn),   *(const v8bf*)(w1 + kn + 8));
    v16bf b2n  = cat8(*(const v8bf*)(w2 + kn),   *(const v8bf*)(w2 + kn + 8));
    v16bf b3n  = cat8(*(const v8bf*)(w3 + kn),   *(const v8bf*)(w3 + kn + 8));
    acc0 = wmma_bf16(af, bf0, acc0);
    acc1 = wmma_bf16(af, bf1, acc1);
    acc2 = wmma_bf16(af, bf2, acc2);
    acc3 = wmma_bf16(af, bf3, acc3);
    af = afn; bf0 = b0n; bf1 = b1n; bf2 = b2n; bf3 = b3n;
  }
  acc0 = wmma_bf16(af, bf0, acc0);
  acc1 = wmma_bf16(af, bf1, acc1);
  acc2 = wmma_bf16(af, bf2, acc2);
  acc3 = wmma_bf16(af, bf3, acc3);

  v8f accs[4] = {acc0, acc1, acc2, acc3};
#pragma unroll
  for (int t = 0; t < 4; ++t) {
    const int n = nbase + t * 16 + l16;
    const float bv = bias[n];
#pragma unroll
    for (int r = 0; r < 8; ++r) {
      const int m = mbase + r + half * 8;
      C[(size_t)m * D_ + n] = accs[t][r] + bv;
    }
  }
}

// ---------------- RoPE + layout change + V transpose ----------------
// thread per (b,h,t,d) with d in [0,32): pairs (d, d+32)
__global__ void rope_transpose_kernel(
    const float* __restrict__ qf, const float* __restrict__ kf,
    const float* __restrict__ vf, const float* __restrict__ cosb,
    const float* __restrict__ sinb, bf16_t* __restrict__ qbf,
    bf16_t* __restrict__ kbf, bf16_t* __restrict__ vtb) {
  int idx = blockIdx.x * blockDim.x + threadIdx.x;   // B*H*T*32 = 2^21
  int d  = idx & 31;
  int t  = (idx >> 5) & (T_ - 1);
  int hh = (idx >> 16) & (H_ - 1);
  int b  = idx >> 20;
  if (b >= B_) return;
  size_t src = ((size_t)b * T_ + t) * D_ + hh * DH_ + d;
  float c = cosb[t * 32 + d], s = sinb[t * 32 + d];
  float q1 = qf[src], q2 = qf[src + 32];
  float k1 = kf[src], k2 = kf[src + 32];
  size_t dst = (((size_t)b * H_ + hh) * T_ + t) * DH_ + d;
  qbf[dst]      = (bf16_t)(q1 * c - q2 * s);
  qbf[dst + 32] = (bf16_t)(q1 * s + q2 * c);
  kbf[dst]      = (bf16_t)(k1 * c - k2 * s);
  kbf[dst + 32] = (bf16_t)(k1 * s + k2 * c);
  size_t vdst = (((size_t)b * H_ + hh) * DH_ + d) * T_ + t;   // [B,H,DH,T]
  vtb[vdst]            = (bf16_t)vf[src];
  vtb[vdst + 32 * T_]  = (bf16_t)vf[src + 32];
}

// ---------------- gate = sigmoid(x[:, :20] @ Wg.T + bg) ----------------
__global__ void gate_kernel(const float* __restrict__ x,
                            const float* __restrict__ Wg,
                            const float* __restrict__ bg,
                            float* __restrict__ gate) {
  int idx = blockIdx.x * blockDim.x + threadIdx.x;   // M_*H_ = 65536
  int h = idx & (H_ - 1);
  int bt = idx >> 4;
  if (bt >= M_) return;
  const float* xr = x + (size_t)bt * D_;
  const float* wr = Wg + h * 20;
  float a = bg[h];
#pragma unroll
  for (int g = 0; g < 20; ++g) a += xr[g] * wr[g];
  gate[idx] = 1.0f / (1.0f + __expf(-a));
}

// ---------------- flash sliding-window attention ----------------
// one wave per (b, h, 16-query tile); 32-key chunks.
// K fragments for chunk j+32 prefetch while chunk j computes; V fragments
// load before the softmax VALU block so they hide behind exp/shuffles.
__global__ __launch_bounds__(256) void attn_kernel(
    const bf16_t* __restrict__ qb, const bf16_t* __restrict__ kb,
    const bf16_t* __restrict__ vt, const float* __restrict__ gate,
    bf16_t* __restrict__ obf) {
  __shared__ __align__(16) bf16_t plds[8][16][32];  // per-wave P bounce
  const int lane = threadIdx.x & 31;
  const int wv   = threadIdx.x >> 5;
  const int flat = blockIdx.x * 8 + wv;     // 4096 waves total
  const int qt = flat & 127;
  const int bh = flat >> 7;
  const int h  = bh & (H_ - 1);
  const int b  = bh >> 4;
  const int half = lane >> 4, l16 = lane & 15;
  const int qbase = qt * 16;
  const size_t bhT = ((size_t)b * H_ + h) * T_;

  // Q fragments (A-layout), kept in regs for the whole wave
  const bf16_t* qrow = qb + (bhT + qbase + l16) * DH_ + half * 16;
  const v16bf aq0 = cat8(*(const v8bf*)(qrow),      *(const v8bf*)(qrow + 8));
  const v16bf aq1 = cat8(*(const v8bf*)(qrow + 32), *(const v8bf*)(qrow + 40));

  float mrow[8], lrow[8];
#pragma unroll
  for (int r = 0; r < 8; ++r) { mrow[r] = 0.0f; lrow[r] = 0.0f; }
  v8f oa[4];
#pragma unroll
  for (int t = 0; t < 4; ++t) oa[t] = vzero8();

  int jstart = qbase - WIN_; if (jstart < 0) jstart = 0; jstart &= ~31;

  // K fragment loader for a 32-key chunk (clamped; safe past the end)
  auto loadK = [&](int jc, v16bf* kf) {
    int j0 = jc + l16;      if (j0 > T_ - 1) j0 = T_ - 1;
    int j1 = jc + 16 + l16; if (j1 > T_ - 1) j1 = T_ - 1;
    const bf16_t* kr0 = kb + (bhT + j0) * DH_ + half * 16;
    const bf16_t* kr1 = kb + (bhT + j1) * DH_ + half * 16;
    kf[0] = cat8(*(const v8bf*)(kr0),      *(const v8bf*)(kr0 + 8));
    kf[1] = cat8(*(const v8bf*)(kr0 + 32), *(const v8bf*)(kr0 + 40));
    kf[2] = cat8(*(const v8bf*)(kr1),      *(const v8bf*)(kr1 + 8));
    kf[3] = cat8(*(const v8bf*)(kr1 + 32), *(const v8bf*)(kr1 + 40));
  };

  v16bf kf[4];
  loadK(jstart, kf);

  for (int jc = jstart; jc < qbase + 16; jc += 32) {
    v8f s0 = vzero8(), s1 = vzero8();
    s0 = wmma_bf16(aq0, kf[0], s0);
    s0 = wmma_bf16(aq1, kf[1], s0);
    s1 = wmma_bf16(aq0, kf[2], s1);
    s1 = wmma_bf16(aq1, kf[3], s1);

    // prefetch next chunk's K fragments (overlaps softmax below)
    v16bf kfn[4];
    loadK(jc + 32, kfn);

    // V fragments for this chunk (overlap softmax below)
    int jj = jc + half * 16; if (jj > T_ - 16) jj = T_ - 16;
    v16bf bvf[4];
#pragma unroll
    for (int t = 0; t < 4; ++t) {
      const bf16_t* vr = vt + (((size_t)b * H_ + h) * DH_ + t * 16 + l16) * T_ + jj;
      bvf[t] = cat8(*(const v8bf*)(vr), *(const v8bf*)(vr + 8));
    }

    const bool needMask = !((jc + 31 <= qbase) && (jc >= qbase + 15 - WIN_));
    float p0[8], p1[8];
#pragma unroll
    for (int r = 0; r < 8; ++r) {
      float a0 = s0[r] * 0.125f;
      float a1 = s1[r] * 0.125f;
      if (needMask) {
        const int i  = qbase + r + half * 8;
        const int ja = jc + l16;
        const int jb = jc + 16 + l16;
        if (ja > i || ja < i - WIN_) a0 = -1e30f;
        if (jb > i || jb < i - WIN_) a1 = -1e30f;
      }
      p0[r] = a0; p1[r] = a1;
    }
    // row max across the 16 lanes of each half-wave
    float mloc[8];
#pragma unroll
    for (int r = 0; r < 8; ++r) mloc[r] = fmaxf(p0[r], p1[r]);
#pragma unroll
    for (int off = 1; off < 16; off <<= 1)
#pragma unroll
      for (int r = 0; r < 8; ++r)
        mloc[r] = fmaxf(mloc[r], __shfl_xor(mloc[r], off, 32));
    float alpha[8];
#pragma unroll
    for (int r = 0; r < 8; ++r) {
      const float mn = fmaxf(mrow[r], mloc[r]);   // base 0 -> masked exp == 0
      alpha[r] = __expf(mrow[r] - mn);
      mrow[r] = mn;
      p0[r] = __expf(p0[r] - mn);
      p1[r] = __expf(p1[r] - mn);
    }
    float sloc[8];
#pragma unroll
    for (int r = 0; r < 8; ++r) sloc[r] = p0[r] + p1[r];
#pragma unroll
    for (int off = 1; off < 16; off <<= 1)
#pragma unroll
      for (int r = 0; r < 8; ++r)
        sloc[r] += __shfl_xor(sloc[r], off, 32);
#pragma unroll
    for (int r = 0; r < 8; ++r) lrow[r] = lrow[r] * alpha[r] + sloc[r];
#pragma unroll
    for (int t = 0; t < 4; ++t)
#pragma unroll
      for (int r = 0; r < 8; ++r) oa[t][r] *= alpha[r];

    // P (C-layout f32) -> LDS -> A-layout bf16 (same-wave, DS in-order)
#pragma unroll
    for (int r = 0; r < 8; ++r) {
      plds[wv][r + half * 8][l16]      = (bf16_t)p0[r];
      plds[wv][r + half * 8][16 + l16] = (bf16_t)p1[r];
    }
    const bf16_t* prow = &plds[wv][l16][half * 16];
    const v16bf pa = cat8(*(const v8bf*)(prow), *(const v8bf*)(prow + 8));

    // O += P @ V
#pragma unroll
    for (int t = 0; t < 4; ++t) oa[t] = wmma_bf16(pa, bvf[t], oa[t]);

#pragma unroll
    for (int t = 0; t < 4; ++t) kf[t] = kfn[t];
  }

  // epilogue: O * gate / l  -> bf16 [B,T,DIM]
#pragma unroll
  for (int r = 0; r < 8; ++r) {
    const int i = qbase + r + half * 8;
    const float f = gate[((size_t)b * T_ + i) * H_ + h] / lrow[r];
#pragma unroll
    for (int t = 0; t < 4; ++t)
      obf[((size_t)b * T_ + i) * D_ + h * DH_ + t * 16 + l16] =
          (bf16_t)(oa[t][r] * f);
  }
}

extern "C" void kernel_launch(void* const* d_in, const int* in_sizes, int n_in,
                              void* d_out, int out_size, void* d_ws, size_t ws_size,
                              hipStream_t stream) {
  (void)in_sizes; (void)n_in; (void)out_size; (void)ws_size;
  const float* x    = (const float*)d_in[0];
  const float* Wq   = (const float*)d_in[1];
  const float* bq   = (const float*)d_in[2];
  const float* Wk   = (const float*)d_in[3];
  const float* bk   = (const float*)d_in[4];
  const float* Wv   = (const float*)d_in[5];
  const float* bv   = (const float*)d_in[6];
  const float* Wo   = (const float*)d_in[7];
  const float* bo   = (const float*)d_in[8];
  const float* Wg   = (const float*)d_in[9];
  const float* bg   = (const float*)d_in[10];
  const float* cosb = (const float*)d_in[11];
  const float* sinb = (const float*)d_in[12];
  float* out = (float*)d_out;

  const size_t nX = (size_t)M_ * D_;   // 4,194,304
  const size_t nW = (size_t)D_ * D_;   // 1,048,576
  char* p = (char*)d_ws;
  bf16_t* x_bf  = (bf16_t*)p; p += nX * 2;
  bf16_t* Wq_bf = (bf16_t*)p; p += nW * 2;
  bf16_t* Wk_bf = (bf16_t*)p; p += nW * 2;
  bf16_t* Wv_bf = (bf16_t*)p; p += nW * 2;
  bf16_t* Wo_bf = (bf16_t*)p; p += nW * 2;
  float*  q_f   = (float*)p;  p += nX * 4;
  float*  k_f   = (float*)p;  p += nX * 4;
  float*  v_f   = (float*)p;  p += nX * 4;
  bf16_t* q_bf  = (bf16_t*)p; p += nX * 2;
  bf16_t* k_bf  = (bf16_t*)p; p += nX * 2;
  bf16_t* v_t   = (bf16_t*)p; p += nX * 2;
  float*  gatep = (float*)p;  p += (size_t)M_ * H_ * 4;
  bf16_t* o_bf  = (bf16_t*)p; p += nX * 2;

  f2bf_kernel<<<(int)((nX + 255) / 256), 256, 0, stream>>>(x,  x_bf,  (int)nX);
  f2bf_kernel<<<(int)((nW + 255) / 256), 256, 0, stream>>>(Wq, Wq_bf, (int)nW);
  f2bf_kernel<<<(int)((nW + 255) / 256), 256, 0, stream>>>(Wk, Wk_bf, (int)nW);
  f2bf_kernel<<<(int)((nW + 255) / 256), 256, 0, stream>>>(Wv, Wv_bf, (int)nW);
  f2bf_kernel<<<(int)((nW + 255) / 256), 256, 0, stream>>>(Wo, Wo_bf, (int)nW);

  gemm_bf16_kernel<<<512, 256, 0, stream>>>(x_bf, Wq_bf, bq, q_f);
  gemm_bf16_kernel<<<512, 256, 0, stream>>>(x_bf, Wk_bf, bk, k_f);
  gemm_bf16_kernel<<<512, 256, 0, stream>>>(x_bf, Wv_bf, bv, v_f);

  rope_transpose_kernel<<<(B_ * H_ * T_ * 32) / 256, 256, 0, stream>>>(
      q_f, k_f, v_f, cosb, sinb, q_bf, k_bf, v_t);
  gate_kernel<<<(M_ * H_) / 256, 256, 0, stream>>>(x, Wg, bg, gatep);

  attn_kernel<<<512, 256, 0, stream>>>(q_bf, k_bf, v_t, gatep, o_bf);

  gemm_bf16_kernel<<<512, 256, 0, stream>>>(o_bf, Wo_bf, bo, out);
}
